// RefineNetLoss_70738111365279
// MI455X (gfx1250) — compile-verified, matrix operand order checked by reading, not run on hardware
//
#include <hip/hip_runtime.h>
#include <hip/hip_bf16.h>
#include <math.h>

// ---------------- problem constants (match reference) ----------------
#define B_   4
#define C_   27
#define H_   160
#define W_   160
#define N_   (H_*W_)          // 25600
#define D_   768
#define KIDS 24
#define CPAD 32
#define MIN_PIX 25.0f

#define LS_      0.2f
#define THING_W_ 2.0f
#define L_DB_    0.1f
#define TWO_S2_  0.005f       // 2*sigma^2, sigma=0.05
#define INV_TEMP_ 10.0f
#define L_DISTILL_ 1.0f
#define L_BPL_   0.5f
#define L_ALIGN_ 5.0f
#define L_PROTO_ 0.5f
#define L_ENT_   0.3f
#define L_IU_    0.5f
#define L_IB_    0.5f

// ---------------- workspace layout (float offsets) ----------------
#define OFF_ACC   0            // 32 scalar accumulators
#define OFF_IUC   32           // B*KIDS counts            (96)
#define OFF_IUS1  128          // B*KIDS*C  sum p          (2592)
#define OFF_IUS2  2720         // B*KIDS*C  sum p^2        (2592)
#define OFF_LBL   5312         // B*N int32 labels         (102400)
#define OFF_RN    107712       // B*N f32 1/max(norm,eps)  (102400)
#define OFF_PROBS 210112       // B*C*N f32 probs          (2764800)
#define OFF_S1    2974912      // B*D*CPAD f32             (98304)
#define OFF_M     3073216      // B*D*CPAD f32             (98304)
#define OFF_PT    3171520      // B*C*N bf16 (ushort)      (1382400 f32 slots)
#define OFF_Q     4553920      // B*C*N bf16 (ushort)      (1382400 f32 slots)
// total ~5.94M floats ~ 23.8 MB

// ACC indices
#define A_GSUM  0
#define A_DIST  1
#define A_BPLN  2
#define A_BPLD  3
#define A_ENT   4
#define A_ALH   5
#define A_ALV   6
#define A_IBH   7
#define A_IBW   8
#define A_PROTO 9
#define A_IUN   10
#define A_IUC   11

typedef __attribute__((ext_vector_type(16))) __bf16 v16bf;
typedef __attribute__((ext_vector_type(8)))  float  v8f;

__device__ __forceinline__ unsigned short f2bf(float f) {
  unsigned u = __float_as_uint(f);
  unsigned r = 0x7FFFu + ((u >> 16) & 1u);
  return (unsigned short)((u + r) >> 16);
}

// pack two f32 into bf16x2 with one v_perm_b32 (truncate): dst = {hi16(y), hi16(x)}
__device__ __forceinline__ unsigned pkbf(float x, float y) {
  return __builtin_amdgcn_perm(__float_as_uint(y), __float_as_uint(x), 0x07060302u);
}

__device__ __forceinline__ float block_reduce256(float v, float* red) {
  int t = threadIdx.x;
  red[t] = v; __syncthreads();
  for (int s = 128; s > 0; s >>= 1) {
    if (t < s) red[t] += red[t + s];
    __syncthreads();
  }
  float r = red[0]; __syncthreads();
  return r;
}

// ---------------- kernel 0: zero accumulators / IU / GEMM outputs ----------------
__global__ void k_zero(float* ws) {
  int idx = blockIdx.x * blockDim.x + threadIdx.x;
  for (int i = idx; i < 5312; i += gridDim.x * blockDim.x) ws[i] = 0.0f;
  for (int i = idx; i < 2 * B_ * D_ * CPAD; i += gridDim.x * blockDim.x)
    ws[OFF_S1 + i] = 0.0f;   // covers S1 then M (contiguous)
}

// ---------------- kernel 1: cause argmax labels + sum of depth-gradient magnitude ----------------
__global__ __launch_bounds__(256) void k_labels(const float* __restrict__ cause,
                                                const float* __restrict__ depth,
                                                float* ws) {
  __shared__ float red[256];
  int b = blockIdx.x / (N_ / 256);
  int n = (blockIdx.x % (N_ / 256)) * 256 + threadIdx.x;
  const float* cb = cause + (size_t)b * C_ * N_ + n;
  float best = cb[0]; int bi = 0;
  #pragma unroll
  for (int c = 1; c < C_; ++c) {
    float v = cb[(size_t)c * N_];
    if (v > best) { best = v; bi = c; }
  }
  ((int*)(ws + OFF_LBL))[b * N_ + n] = bi;

  int h = n / W_, w = n % W_;
  const float* db = depth + (size_t)b * N_;
  float d0 = db[n];
  float gh = (h < H_ - 1) ? fabsf(db[n + W_] - d0) : 0.0f;
  float gw = (w < W_ - 1) ? fabsf(db[n + 1] - d0) : 0.0f;
  float g = block_reduce256(gh + gw, red);
  if (threadIdx.x == 0) atomicAdd(&ws[OFF_ACC + A_GSUM], g);
}

// ---------------- kernel 2: per-pixel dinov2 feature norms (4 pixels/thread, b128) ----------------
__global__ __launch_bounds__(256) void k_norms(const float* __restrict__ feats, float* ws) {
  int b  = blockIdx.x / (N_ / 1024);
  int n4 = (blockIdx.x % (N_ / 1024)) * 1024 + threadIdx.x * 4;
  const float* fb = feats + (size_t)b * D_ * N_ + n4;
  float sx = 0.f, sy = 0.f, sz = 0.f, sw = 0.f;
  #pragma unroll 4
  for (int d = 0; d < D_; ++d) {
    float4 v = *(const float4*)(fb + (size_t)d * N_);
    sx = fmaf(v.x, v.x, sx);
    sy = fmaf(v.y, v.y, sy);
    sz = fmaf(v.z, v.z, sz);
    sw = fmaf(v.w, v.w, sw);
  }
  float4 r;
  r.x = 1.0f / fmaxf(sqrtf(sx), 1e-12f);
  r.y = 1.0f / fmaxf(sqrtf(sy), 1e-12f);
  r.z = 1.0f / fmaxf(sqrtf(sz), 1e-12f);
  r.w = 1.0f / fmaxf(sqrtf(sw), 1e-12f);
  *(float4*)(ws + OFF_RN + b * N_ + n4) = r;
}

// ---------------- kernel 3: softmax pass: probs f32, probs_t & q bf16 ----------------
__global__ __launch_bounds__(256) void k_softmax(const float* __restrict__ logits, float* ws) {
  int b = blockIdx.x / (N_ / 256);
  int n = (blockIdx.x % (N_ / 256)) * 256 + threadIdx.x;
  const float* lb = logits + (size_t)b * C_ * N_ + n;
  float l[C_];
  float m = -1e30f;
  #pragma unroll
  for (int c = 0; c < C_; ++c) { l[c] = lb[(size_t)c * N_]; m = fmaxf(m, l[c]); }
  float s = 0.0f, sT = 0.0f;
  #pragma unroll
  for (int c = 0; c < C_; ++c) {
    s  += expf(l[c] - m);
    sT += expf((l[c] - m) * INV_TEMP_);
  }
  float inv = 1.0f / s, invT = 1.0f / sT;
  float rn = ws[OFF_RN + b * N_ + n];
  float* pr = ws + OFF_PROBS + (size_t)b * C_ * N_ + n;
  unsigned short* pt = (unsigned short*)(ws + OFF_PT) + (size_t)b * C_ * N_ + n;
  unsigned short* qb = (unsigned short*)(ws + OFF_Q)  + (size_t)b * C_ * N_ + n;
  #pragma unroll
  for (int c = 0; c < C_; ++c) {
    pr[(size_t)c * N_] = expf(l[c] - m) * inv;
    float p = expf((l[c] - m) * INV_TEMP_) * invT;
    pt[(size_t)c * N_] = f2bf(p);
    qb[(size_t)c * N_] = f2bf(p * rn);
  }
}

// ---------------- kernel 4: dual-output bf16 WMMA GEMM over feats ----------------
// S1[b,d,c] = sum_n feats[b,d,n]*probs_t[b,c,n]
// M [b,d,c] = sum_n feats[b,d,n]*q[b,c,n]
// grid = B * 12(d-tiles of 64) * 8(K-chunks); block = 128 threads (4 waves).
// Fragment layouts partition tiles exactly across lanes: A and B are loaded
// DIRECTLY from global (no LDS, no barriers). Columns 27..31 are clamped to 26
// instead of masked — S1/M cols >=27 are never read, and WMMA output columns
// are independent, so no divergence is needed in the hot loop.
#define KSPLIT 8
__global__ __launch_bounds__(128) void k_gemm(const float* __restrict__ feats, float* ws) {
  int blk = blockIdx.x;
  int b   = blk / (12 * KSPLIT);
  int rem = blk % (12 * KSPLIT);
  int dt  = rem / KSPLIT;
  int ks  = rem % KSPLIT;
  int d0  = dt * 64;

  int tid = threadIdx.x, lane = tid & 31, wave = tid >> 5;
  int lrow = lane & 15, hf = lane >> 4;

  // A: this lane owns row (d0 + wave*16 + lrow), K elements hf*8..+7 and 16+hf*8..+7
  const float* asrc = feats + ((size_t)b * D_ + d0 + wave * 16 + lrow) * N_;

  // B: lane = column; [c][n] bf16 layout in ws is exactly fragment-contiguous
  const unsigned short* ptb = (const unsigned short*)(ws + OFF_PT) + (size_t)b * C_ * N_;
  const unsigned short* qqb = (const unsigned short*)(ws + OFF_Q)  + (size_t)b * C_ * N_;
  int col0 = lrow;                                   // 0..15, always valid
  int col1 = (16 + lrow < C_) ? (16 + lrow) : C_ - 1; // clamp 27..31 -> 26 (outputs unused)
  const unsigned short* bp00 = ptb + (size_t)col0 * N_ + hf * 8;
  const unsigned short* bp01 = ptb + (size_t)col1 * N_ + hf * 8;
  const unsigned short* bp10 = qqb + (size_t)col0 * N_ + hf * 8;
  const unsigned short* bp11 = qqb + (size_t)col1 * N_ + hf * 8;

  v8f zero = {0.f, 0.f, 0.f, 0.f, 0.f, 0.f, 0.f, 0.f};
  v8f acc00 = zero, acc01 = zero, acc10 = zero, acc11 = zero;

  int n0 = ks * (N_ / KSPLIT);
  const int iters = (N_ / KSPLIT) / 32;   // 100

  union U { v16bf v; unsigned u[8]; uint4 q[2]; };

  for (int it = 0; it < iters; ++it, n0 += 32) {
    int ka = n0 + hf * 8;
    float4 a0 = *(const float4*)(asrc + ka);
    float4 a1 = *(const float4*)(asrc + ka + 4);
    float4 a2 = *(const float4*)(asrc + ka + 16);
    float4 a3 = *(const float4*)(asrc + ka + 20);

    U b00, b01, b10, b11;
    b00.q[0] = *(const uint4*)(bp00 + n0);  b00.q[1] = *(const uint4*)(bp00 + n0 + 16);
    b01.q[0] = *(const uint4*)(bp01 + n0);  b01.q[1] = *(const uint4*)(bp01 + n0 + 16);
    b10.q[0] = *(const uint4*)(bp10 + n0);  b10.q[1] = *(const uint4*)(bp10 + n0 + 16);
    b11.q[0] = *(const uint4*)(bp11 + n0);  b11.q[1] = *(const uint4*)(bp11 + n0 + 16);

    U a;
    a.u[0] = pkbf(a0.x, a0.y); a.u[1] = pkbf(a0.z, a0.w);
    a.u[2] = pkbf(a1.x, a1.y); a.u[3] = pkbf(a1.z, a1.w);
    a.u[4] = pkbf(a2.x, a2.y); a.u[5] = pkbf(a2.z, a2.w);
    a.u[6] = pkbf(a3.x, a3.y); a.u[7] = pkbf(a3.z, a3.w);

    acc00 = __builtin_amdgcn_wmma_f32_16x16x32_bf16(false, a.v, false, b00.v, (short)0, acc00, false, false);
    acc01 = __builtin_amdgcn_wmma_f32_16x16x32_bf16(false, a.v, false, b01.v, (short)0, acc01, false, false);
    acc10 = __builtin_amdgcn_wmma_f32_16x16x32_bf16(false, a.v, false, b10.v, (short)0, acc10, false, false);
    acc11 = __builtin_amdgcn_wmma_f32_16x16x32_bf16(false, a.v, false, b11.v, (short)0, acc11, false, false);
  }

  // C/D layout: lanes 0-15 col=lane, row=r; lanes 16-31 row=r+8
  float* S1o = ws + OFF_S1 + (size_t)b * D_ * CPAD;
  float* Mo  = ws + OFF_M  + (size_t)b * D_ * CPAD;
  #pragma unroll
  for (int r = 0; r < 8; ++r) {
    int drow = d0 + wave * 16 + (hf ? r + 8 : r);
    atomicAdd(&S1o[drow * CPAD + lrow],      acc00[r]);
    atomicAdd(&S1o[drow * CPAD + 16 + lrow], acc01[r]);
    atomicAdd(&Mo [drow * CPAD + lrow],      acc10[r]);
    atomicAdd(&Mo [drow * CPAD + 16 + lrow], acc11[r]);
  }
}

// ---------------- kernel 5: main per-pixel losses + IU histogram ----------------
__global__ __launch_bounds__(256) void k_main(const float* __restrict__ depth,
                                              const int* __restrict__ inst,
                                              float* ws) {
  __shared__ float red[256];
  __shared__ float iu[KIDS * (1 + 2 * C_)];   // [counts | s1 | s2]
  int b = blockIdx.x / (N_ / 256);
  int n = (blockIdx.x % (N_ / 256)) * 256 + threadIdx.x;

  for (int i = threadIdx.x; i < KIDS * (1 + 2 * C_); i += 256) iu[i] = 0.0f;
  __syncthreads();

  const float* pr  = ws + OFF_PROBS + (size_t)b * C_ * N_;
  const int*   lbl = (const int*)(ws + OFF_LBL) + b * N_;
  const float* db  = depth + (size_t)b * N_;
  const int*   ib  = inst + (size_t)b * N_;
  int h = n / W_, w = n % W_;
  int lab = lbl[n];

  float p[C_];
  float sumlog = 0.0f, ent = 0.0f, nll = 0.0f;
  #pragma unroll
  for (int c = 0; c < C_; ++c) {
    p[c] = pr[(size_t)c * N_ + n];
    float lp = logf(p[c]);
    sumlog += lp;
    ent -= p[c] * lp;
    if (c == lab) nll = -lp;            // avoid dynamic index -> keep p[] in VGPRs
  }
  float ce  = (1.0f - LS_) * nll + LS_ * (-sumlog / (float)C_);
  float pt  = expf(-ce);
  float wgt = (1.0f - pt) * (1.0f - pt);
  if (lab >= 11 && lab <= 18) wgt *= THING_W_;

  float dd0 = db[n];
  float gh = (h < H_ - 1) ? fabsf(db[n + W_] - dd0) : 0.0f;
  float gw = (w < W_ - 1) ? fabsf(db[n + 1] - dd0) : 0.0f;
  float gmean = ws[OFF_ACC + A_GSUM] / (float)(B_ * N_);
  float gn = (gh + gw) / (gmean + 1e-6f);
  wgt *= (1.0f + L_DB_ * fminf(gn, 3.0f));
  float a_dist = ce * wgt;

  int lr = lbl[(w < W_ - 1) ? n + 1 : n];
  int ll = lbl[(w > 0)      ? n - 1 : n];
  int ld = lbl[(h < H_ - 1) ? n + W_ : n];
  int lu = lbl[(h > 0)      ? n - W_ : n];
  float bnd = (lab != lr || lab != ll || lab != ld || lab != lu) ? 1.0f : 0.0f;

  float a_ah = 0.0f, a_av = 0.0f, a_ibh = 0.0f, a_ibw = 0.0f;
  if (w < W_ - 1) {
    float ddh = fabsf(db[n + 1] - dd0);
    float whr = expf(-(ddh * ddh) / TWO_S2_);
    float pd = 0.0f, mx = 0.0f;
    #pragma unroll
    for (int c = 0; c < C_; ++c) {
      float q = pr[(size_t)c * N_ + n + 1] - p[c];
      pd = fmaf(q, q, pd);
      mx = fmaxf(mx, fabsf(q));
    }
    a_ah = whr * pd;
    float y = (ib[n + 1] != ib[n]) ? 1.0f : 0.0f;
    float x = fminf(fmaxf(mx, 1e-6f), 1.0f - 1e-6f);
    a_ibw = -(y * logf(x) + (1.0f - y) * log1pf(-x));
  }
  if (h < H_ - 1) {
    float ddv = fabsf(db[n + W_] - dd0);
    float wvr = expf(-(ddv * ddv) / TWO_S2_);
    float pd = 0.0f, mx = 0.0f;
    #pragma unroll
    for (int c = 0; c < C_; ++c) {
      float q = pr[(size_t)c * N_ + n + W_] - p[c];
      pd = fmaf(q, q, pd);
      mx = fmaxf(mx, fabsf(q));
    }
    a_av = wvr * pd;
    float y = (ib[n + W_] != ib[n]) ? 1.0f : 0.0f;
    float x = fminf(fmaxf(mx, 1e-6f), 1.0f - 1e-6f);
    a_ibh = -(y * logf(x) + (1.0f - y) * log1pf(-x));
  }

  // instance-uniformity stats into LDS (ds_add_f32)
  int id = ib[n];
  atomicAdd(&iu[id], 1.0f);
  #pragma unroll
  for (int c = 0; c < C_; ++c) {
    atomicAdd(&iu[KIDS + id * C_ + c], p[c]);
    atomicAdd(&iu[KIDS + KIDS * C_ + id * C_ + c], p[c] * p[c]);
  }

  float r;
  r = block_reduce256(a_dist, red);   if (threadIdx.x == 0) atomicAdd(&ws[OFF_ACC + A_DIST], r);
  r = block_reduce256(ce * bnd, red); if (threadIdx.x == 0) atomicAdd(&ws[OFF_ACC + A_BPLN], r);
  r = block_reduce256(bnd, red);      if (threadIdx.x == 0) atomicAdd(&ws[OFF_ACC + A_BPLD], r);
  r = block_reduce256(ent, red);      if (threadIdx.x == 0) atomicAdd(&ws[OFF_ACC + A_ENT], r);
  r = block_reduce256(a_ah, red);     if (threadIdx.x == 0) atomicAdd(&ws[OFF_ACC + A_ALH], r);
  r = block_reduce256(a_av, red);     if (threadIdx.x == 0) atomicAdd(&ws[OFF_ACC + A_ALV], r);
  r = block_reduce256(a_ibh, red);    if (threadIdx.x == 0) atomicAdd(&ws[OFF_ACC + A_IBH], r);
  r = block_reduce256(a_ibw, red);    if (threadIdx.x == 0) atomicAdd(&ws[OFF_ACC + A_IBW], r);
  __syncthreads();

  for (int i = threadIdx.x; i < KIDS; i += 256)
    atomicAdd(&ws[OFF_IUC + b * KIDS + i], iu[i]);
  for (int i = threadIdx.x; i < KIDS * C_; i += 256) {
    atomicAdd(&ws[OFF_IUS1 + b * KIDS * C_ + i], iu[KIDS + i]);
    atomicAdd(&ws[OFF_IUS2 + b * KIDS * C_ + i], iu[KIDS + KIDS * C_ + i]);
  }
}

// ---------------- kernel 6: IU finalize ----------------
__global__ __launch_bounds__(32) void k_iu(float* ws) {
  __shared__ float red[32];
  int b = blockIdx.x / KIDS, k = blockIdx.x % KIDS;
  int c = threadIdx.x;
  float cnt = ws[OFF_IUC + b * KIDS + k];
  float v = 0.0f;
  if (c < C_) {
    float nc = fmaxf(cnt, 1.0f);
    float s1 = ws[OFF_IUS1 + (b * KIDS + k) * C_ + c];
    float s2 = ws[OFF_IUS2 + (b * KIDS + k) * C_ + c];
    float mu = s1 / nc;
    v = s2 / nc - mu * mu;
  }
  red[c] = v; __syncthreads();
  for (int s = 16; s > 0; s >>= 1) {
    if (c < s) red[c] += red[c + s];
    __syncthreads();
  }
  if (c == 0) {
    float var = red[0] / (float)C_;
    if (k > 0 && cnt >= MIN_PIX) {
      atomicAdd(&ws[OFF_ACC + A_IUN], var);
      atomicAdd(&ws[OFF_ACC + A_IUC], 1.0f);
    }
  }
}

// ---------------- kernel 7: proto finalize (per (b,c) column dot & norm) ----------------
__global__ __launch_bounds__(256) void k_proto(float* ws) {
  __shared__ float rd[256];
  __shared__ float rn[256];
  int b = blockIdx.x / C_, c = blockIdx.x % C_;
  const float* S1 = ws + OFF_S1 + (size_t)b * D_ * CPAD;
  const float* M  = ws + OFF_M  + (size_t)b * D_ * CPAD;
  float dot = 0.0f, nr = 0.0f;
  for (int d = threadIdx.x; d < D_; d += 256) {
    float s = S1[d * CPAD + c];
    float m = M[d * CPAD + c];
    dot = fmaf(s, m, dot);
    nr  = fmaf(s, s, nr);
  }
  int t = threadIdx.x;
  rd[t] = dot; rn[t] = nr; __syncthreads();
  for (int s = 128; s > 0; s >>= 1) {
    if (t < s) { rd[t] += rd[t + s]; rn[t] += rn[t + s]; }
    __syncthreads();
  }
  if (t == 0)
    atomicAdd(&ws[OFF_ACC + A_PROTO], rd[0] / fmaxf(sqrtf(rn[0]), 1e-12f));
}

// ---------------- kernel 8: final scalar combine ----------------
__global__ void k_final(const float* __restrict__ ws, float* out) {
  const float BN = (float)(B_ * N_);
  float l_distill = ws[OFF_ACC + A_DIST] / BN;
  float bden = ws[OFF_ACC + A_BPLD];
  float l_bpl = (bden > 0.0f) ? ws[OFF_ACC + A_BPLN] / fmaxf(bden, 1.0f) : 0.0f;
  float l_align = ws[OFF_ACC + A_ALH] / (float)(B_ * H_ * (W_ - 1))
                + ws[OFF_ACC + A_ALV] / (float)(B_ * (H_ - 1) * W_);
  float l_proto = -ws[OFF_ACC + A_PROTO] / BN;
  float l_ent = ws[OFF_ACC + A_ENT] / BN;
  float l_ib = 0.5f * (ws[OFF_ACC + A_IBH] / (float)(B_ * (H_ - 1) * W_)
                     + ws[OFF_ACC + A_IBW] / (float)(B_ * H_ * (W_ - 1)));
  float l_iu = ws[OFF_ACC + A_IUN] / fmaxf(ws[OFF_ACC + A_IUC], 1.0f);
  out[0] = L_DISTILL_ * l_distill + L_BPL_ * l_bpl + L_ALIGN_ * l_align
         + L_PROTO_ * l_proto + L_ENT_ * l_ent + L_IU_ * l_iu + L_IB_ * l_ib;
}

// ---------------- launcher ----------------
extern "C" void kernel_launch(void* const* d_in, const int* in_sizes, int n_in,
                              void* d_out, int out_size, void* d_ws, size_t ws_size,
                              hipStream_t stream) {
  const float* logits = (const float*)d_in[0];
  const float* cause  = (const float*)d_in[1];
  const float* feats  = (const float*)d_in[2];
  const float* depth  = (const float*)d_in[3];
  const int*   inst   = (const int*)d_in[4];
  float* ws  = (float*)d_ws;
  float* out = (float*)d_out;

  const int PIX_BLOCKS = B_ * (N_ / 256);   // 400

  k_zero   <<<256, 256, 0, stream>>>(ws);
  k_labels <<<PIX_BLOCKS, 256, 0, stream>>>(cause, depth, ws);
  k_norms  <<<B_ * (N_ / 1024), 256, 0, stream>>>(feats, ws);
  k_softmax<<<PIX_BLOCKS, 256, 0, stream>>>(logits, ws);
  k_gemm   <<<B_ * 12 * KSPLIT, 128, 0, stream>>>(feats, ws);
  k_main   <<<PIX_BLOCKS, 256, 0, stream>>>(depth, inst, ws);
  k_iu     <<<B_ * KIDS, 32, 0, stream>>>(ws);
  k_proto  <<<B_ * C_, 256, 0, stream>>>(ws);
  k_final  <<<1, 1, 0, stream>>>(ws, out);
}
